// MoEGate_44616120271589
// MI455X (gfx1250) — matrix operand order repaired and checked by reading, not compiled, not run on hardware
//
#include <hip/hip_runtime.h>
#include <hip/hip_bf16.h>
#include <stdint.h>

// DeepSeek-V3 MoE gate for MI455X (gfx1250, wave32, WMMA).
#define SEQ    16384
#define HID    7168
#define NEXP   256
#define NGRP   8
#define GSZ    32      // experts per group
#define TOPKG  4
#define TOPK   8
#define KSTAGE 64      // K per LDS stage (two 32-K WMMA sub-steps)
#define NSTAGE (HID / KSTAGE)   // 112
#define TILE_M 32      // rows per workgroup
#define LDA    72      // A-tile LDS stride in ushorts (64 + 8 pad: spreads banks)
#define LDS_E  264     // logits LDS stride in floats (256 + 8 pad)

typedef __attribute__((ext_vector_type(16))) __bf16 v16bf;
typedef __attribute__((ext_vector_type(8)))  float  v8f;

union Frag { uint4 q[2]; v16bf v; };

// fp32 -> bf16 (round half up), pack two into one dword: 2x v_add + 1x v_perm
static __device__ __forceinline__ unsigned int pack2bf(float x, float y) {
    unsigned int ux = __float_as_uint(x) + 0x8000u;
    unsigned int uy = __float_as_uint(y) + 0x8000u;
    // dst = { uy[31:16], ux[31:16] }
    return __builtin_amdgcn_perm(uy, ux, 0x07060302u);
}

// ---------------- kernel 0: weight fp32 -> bf16 (once, into d_ws) -------------
__global__ void __launch_bounds__(256)
wcvt_kernel(const float* __restrict__ w, unsigned int* __restrict__ wbf, int n4) {
    int i = blockIdx.x * 256 + threadIdx.x;
    if (i >= n4) return;
    float4 f = ((const float4*)w)[i];
    uint2 p;
    p.x = pack2bf(f.x, f.y);
    p.y = pack2bf(f.z, f.w);
    ((uint2*)wbf)[i] = p;
}

// ---------------- kernel 1: fused router GEMM + sigmoid + grouped top-k ------
__global__ void __launch_bounds__(256)
moe_gate_kernel(const float* __restrict__ hidden,
                const unsigned short* __restrict__ wbf,   // bf16 weight [NEXP][HID]
                const float* __restrict__ bias,
                float* __restrict__ outW,
                int* __restrict__ outI) {
    __shared__ __align__(16) unsigned short aTile[2 * TILE_M * LDA]; // double buffer
    __shared__ float sTile[TILE_M * LDS_E];
    __shared__ float biasLds[NEXP];

    const int tid     = threadIdx.x;
    const int rowBase = blockIdx.x * TILE_M;
    const int wave    = tid >> 5;
    const int lane    = tid & 31;
    const int eb      = wave * 32;        // each wave owns a unique 32-expert strip

    biasLds[tid] = bias[tid];             // 256 threads == 256 experts

    const v8f vzero = {0.f,0.f,0.f,0.f,0.f,0.f,0.f,0.f};
    v8f acc[2][2];                        // [row-tile mt][expert-tile t]
    #pragma unroll
    for (int m = 0; m < 2; ++m)
        #pragma unroll
        for (int t = 0; t < 2; ++t) acc[m][t] = vzero;

    // cooperative A staging: thread -> (row, 8 cols) of the 32x64 stage tile
    const int sRow = tid >> 3;
    const int sCol = (tid & 7) * 8;
    const float* aSrc = hidden + (long long)(rowBase + sRow) * HID + sCol;
    const int sDst = sRow * LDA + sCol;   // ushort index within one buffer

    // A-fragment LDS addressing (ISA 16-bit A 16x32 layout):
    //   lanes 0-15 (g=0): K {0..7, 16..23};  lanes 16-31 (g=1): K {8..15, 24..31}
    const int am = lane & 15;
    const int ag = lane >> 4;
    // B-fragment addressing (32x16 B layout): N = lane&15, same K order as A: g=0 K {0..7,16..23}, g=1 K {8..15,24..31}
    const int bn = lane & 15;
    const int bk = (lane >> 4) * 8;

    // ---- prologue: stage tile 0 ----
    {
        float4 f0 = *(const float4*)(aSrc);
        float4 f1 = *(const float4*)(aSrc + 4);
        uint4 pk;
        pk.x = pack2bf(f0.x, f0.y);
        pk.y = pack2bf(f0.z, f0.w);
        pk.z = pack2bf(f1.x, f1.y);
        pk.w = pack2bf(f1.z, f1.w);
        *(uint4*)&aTile[sDst] = pk;
    }
    __syncthreads();

    for (int s = 0; s < NSTAGE; ++s) {
        const int k0  = s * KSTAGE;
        const int cur = (s & 1) * (TILE_M * LDA);

        // issue next stage's global loads early (overlap with WMMA below)
        float4 g0, g1;
        if (s + 1 < NSTAGE) {
            g0 = *(const float4*)(aSrc + k0 + KSTAGE);
            g1 = *(const float4*)(aSrc + k0 + KSTAGE + 4);
        }

        #pragma unroll
        for (int ks = 0; ks < 2; ++ks) {
            const int kc = 32 * ks;
            // A fragments for both 16-row tiles
            const unsigned short* ap0 = &aTile[cur + am * LDA + kc + 8 * ag];
            Frag A0, A1;
            A0.q[0] = *(const uint4*)(ap0);
            A0.q[1] = *(const uint4*)(ap0 + 16);
            A1.q[0] = *(const uint4*)(ap0 + 16 * LDA);
            A1.q[1] = *(const uint4*)(ap0 + 16 * LDA + 16);
            #pragma unroll
            for (int t = 0; t < 2; ++t) {
                const unsigned short* bp =
                    wbf + (long long)(eb + t * 16 + bn) * HID + k0 + kc + bk;
                Frag B;
                B.q[0] = *(const uint4*)(bp);       // 8 bf16 at K 8g.., then 8 at K 16+8g..
                B.q[1] = *(const uint4*)(bp + 16);
                acc[0][t] = __builtin_amdgcn_wmma_f32_16x16x32_bf16(
                    false, A0.v, false, B.v, (short)0, acc[0][t], false, false);
                acc[1][t] = __builtin_amdgcn_wmma_f32_16x16x32_bf16(
                    false, A1.v, false, B.v, (short)0, acc[1][t], false, false);
            }
        }

        // pack + store next stage into the other buffer, then one barrier
        if (s + 1 < NSTAGE) {
            const int nxt = ((s + 1) & 1) * (TILE_M * LDA);
            uint4 pk;
            pk.x = pack2bf(g0.x, g0.y);
            pk.y = pack2bf(g0.z, g0.w);
            pk.z = pack2bf(g1.x, g1.y);
            pk.w = pack2bf(g1.z, g1.w);
            *(uint4*)&aTile[nxt + sDst] = pk;
        }
        __syncthreads();
    }

    // write accumulators to LDS (ISA C layout: VGPR v -> M = 8*(lane>=16)+v, N=lane&15)
    {
        const int n = lane & 15;
        const int mb = (lane >> 4) * 8;
        #pragma unroll
        for (int m = 0; m < 2; ++m) {
            #pragma unroll
            for (int t = 0; t < 2; ++t) {
                #pragma unroll
                for (int v = 0; v < 8; ++v) {
                    sTile[(m * 16 + mb + v) * LDS_E + eb + t * 16 + n] = acc[m][t][v];
                }
            }
        }
    }
    __syncthreads();

    // sigmoid in place: each thread owns 32 logits
    {
        const int r  = tid >> 3;
        const int c0 = (tid & 7) * 32;
        #pragma unroll 4
        for (int j = 0; j < 32; ++j) {
            float x = sTile[r * LDS_E + c0 + j];
            sTile[r * LDS_E + c0 + j] = 1.0f / (1.0f + __expf(-x));
        }
    }
    __syncthreads();

    // gating: one thread per row (wave 0, all 32 lanes active)
    if (tid < TILE_M) {
        const float* s = &sTile[tid * LDS_E];

        // group scores: sum of top-2 corrected scores per group
        float gs[NGRP];
        #pragma unroll
        for (int g = 0; g < NGRP; ++g) {
            float m1 = -__builtin_inff(), m2 = -__builtin_inff();
            #pragma unroll 4
            for (int i = 0; i < GSZ; ++i) {
                float c = s[g * GSZ + i] + biasLds[g * GSZ + i];
                if (c > m1)      { m2 = m1; m1 = c; }
                else if (c > m2) { m2 = c; }
            }
            gs[g] = m1 + m2;
        }
        // top-4 groups
        unsigned int gsel = 0;
        for (int p = 0; p < TOPKG; ++p) {
            float best = -__builtin_inff(); int bg = 0;
            #pragma unroll
            for (int g = 0; g < NGRP; ++g)
                if (!((gsel >> g) & 1u) && gs[g] > best) { best = gs[g]; bg = g; }
            gsel |= 1u << bg;
        }
        // masked global top-8 (strict '>' keeps lowest index on ties, like lax.top_k)
        unsigned int picked[8] = {0,0,0,0,0,0,0,0};
        int   idx[TOPK];
        float wt[TOPK];
        float denom = 1e-20f;
        for (int p = 0; p < TOPK; ++p) {
            float best = -__builtin_inff(); int bi = 0;
            #pragma unroll 4
            for (int e = 0; e < NEXP; ++e) {
                if ((picked[e >> 5] >> (e & 31)) & 1u) continue;
                float c = ((gsel >> (e >> 5)) & 1u) ? (s[e] + biasLds[e]) : 0.0f;
                if (c > best) { best = c; bi = e; }
            }
            picked[bi >> 5] |= 1u << (bi & 31);
            idx[p] = bi;
            wt[p]  = s[bi];        // UNCORRECTED sigmoid as weight
            denom += wt[p];
        }
        const float scale = 2.5f / denom;
        const long long ob = (long long)(rowBase + tid) * TOPK;
        #pragma unroll
        for (int p = 0; p < TOPK; ++p) {
            outW[ob + p] = wt[p] * scale;
            outI[ob + p] = idx[p];
        }
    }
}

extern "C" void kernel_launch(void* const* d_in, const int* in_sizes, int n_in,
                              void* d_out, int out_size, void* d_ws, size_t ws_size,
                              hipStream_t stream) {
    const float* hidden = (const float*)d_in[0];
    const float* weight = (const float*)d_in[1];  // [NEXP, HID] fp32
    const float* bias   = (const float*)d_in[2];  // [NEXP] fp32

    // d_ws: bf16 weight, NEXP*HID*2 = 3.67 MB
    unsigned int* wbf = (unsigned int*)d_ws;
    const int n4 = NEXP * HID / 4;                // 458752 float4 groups
    wcvt_kernel<<<n4 / 256, 256, 0, stream>>>(weight, wbf, n4);

    float* outW = (float*)d_out;                       // topk_weight [SEQ,8] fp32
    int*   outI = (int*)d_out + (long long)SEQ * TOPK; // topk_idx    [SEQ,8] i32

    moe_gate_kernel<<<SEQ / TILE_M, 256, 0, stream>>>(
        hidden, (const unsigned short*)d_ws, bias, outW, outI);
}
